// Struct2Seq_68315749810793
// MI455X (gfx1250) — matrix operand order repaired and checked by reading, not compiled, not run on hardware
//
#include <hip/hip_runtime.h>
#include <cstdint>
#include <cstddef>

typedef __attribute__((ext_vector_type(16))) _Float16 v16h;
typedef __attribute__((ext_vector_type(8)))  _Float16 v8h;
typedef __attribute__((ext_vector_type(8)))  float    v8f;

namespace cfg {
constexpr int H = 128, Z = 64, VOCAB = 20, K = 30, B = 8, N = 512, FF = 512;
constexpr int BN = B * N;        // 4096 nodes
constexpr int NE = B * N * K;    // 122880 edges
}

// ---------------------------------------------------------------------------
// CDNA5 async global->LDS copy (ASYNCcnt path). Flip to 0 for portable path.
// ---------------------------------------------------------------------------
#define USE_ASYNC_COPY 1

__device__ __forceinline__ void lds_copy_b128(void* lds_dst, const void* gsrc)
{
#if USE_ASYNC_COPY
    // GLOBAL_LOAD_ASYNC_TO_LDS_B128: VDST = LDS byte address, VADDR = 64b global
    unsigned l = (unsigned)(uintptr_t)lds_dst;  // LDS aperture: low 32 bits = offset
    asm volatile("global_load_async_to_lds_b128 %0, %1, off"
                 :: "v"(l), "v"(gsrc) : "memory");
#else
    *(v8h*)lds_dst = *(const v8h*)gsrc;
#endif
}

__device__ __forceinline__ void lds_copy_wait()
{
#if USE_ASYNC_COPY
    asm volatile("s_wait_asynccnt 0x0" ::: "memory");
#endif
}

// ---------------------------------------------------------------------------
// Tiled WMMA GEMM: C[M,Nd] = act(A[M,Kd] @ W + bias), W given pre-transposed
// as WT[Nd,Kd] (both f16 row-major). Block = 128 threads = 4 wave32 waves.
// Block tile 64x64; each wave computes 32x32 via 2x2 wmma_f32_16x16x32_f16
// with A/B fragment reuse. Requires M % 64 == 0, Nd % 64 == 0, Kd % 32 == 0.
// LDS rows padded to 48 halves (96B) so all fragment loads are ds_load_b128
// with 2-way-max bank spread.
// ---------------------------------------------------------------------------
template <bool RELU, bool WRITE_F32>
__global__ __launch_bounds__(128)
void wmma_gemm_kernel(const _Float16* __restrict__ A,
                      const _Float16* __restrict__ WT,
                      const float* __restrict__ bias,
                      _Float16* __restrict__ C16,
                      float* __restrict__ C32,
                      int M, int Kd, int Nd)
{
    __shared__ _Float16 As[64][48];   // [m][k]
    __shared__ _Float16 Bt[64][48];   // [n][k]

    const int tid  = threadIdx.x;
    const int lane = tid & 31;
    const int wave = tid >> 5;
    const int rowBlk = blockIdx.x * 64;
    const int colBlk = blockIdx.y * 64;
    const int mo = (wave >> 1) * 32;      // wave row offset in 64x64 tile
    const int no = (wave & 1) * 32;       // wave col offset in 64x64 tile

    const int trow = tid >> 1;            // 0..63 tile row for cooperative copy
    const int tcol = (tid & 1) * 16;      // 0 or 16 (halves)

    const int half = lane >> 4;           // 0: lanes 0-15, 1: lanes 16-31
    const int l15  = lane & 15;

    v8f acc00 = {}, acc01 = {}, acc10 = {}, acc11 = {};

    const _Float16* aRow = A  + (size_t)(rowBlk + trow) * Kd;
    const _Float16* wRow = WT + (size_t)(colBlk + trow) * Kd;

    for (int k0 = 0; k0 < Kd; k0 += 32) {
        // stage 64x32 A tile and 64x32 WT tile (16 halves per thread each)
        lds_copy_b128(&As[trow][tcol],     aRow + k0 + tcol);
        lds_copy_b128(&As[trow][tcol + 8], aRow + k0 + tcol + 8);
        lds_copy_b128(&Bt[trow][tcol],     wRow + k0 + tcol);
        lds_copy_b128(&Bt[trow][tcol + 8], wRow + k0 + tcol + 8);
        if (k0 + 32 < Kd) {  // uniform
            __builtin_prefetch(aRow + k0 + 32 + tcol, 0, 0);
        }
        lds_copy_wait();
        __syncthreads();

        // A 16x32 f16 fragment: lane l15 = row, halves {half*8..+7, 16+half*8..+7}
        v8h a0l = *(const v8h*)(&As[mo + l15][half * 8]);
        v8h a0h = *(const v8h*)(&As[mo + l15][16 + half * 8]);
        v8h a1l = *(const v8h*)(&As[mo + 16 + l15][half * 8]);
        v8h a1h = *(const v8h*)(&As[mo + 16 + l15][16 + half * 8]);
        // B 32x16 f16 fragment: lane l15 = col, halves {half*8..+7, 16+half*8..+7}
        v8h b0l = *(const v8h*)(&Bt[no + l15][half * 8]);
        v8h b0h = *(const v8h*)(&Bt[no + l15][16 + half * 8]);
        v8h b1l = *(const v8h*)(&Bt[no + 16 + l15][half * 8]);
        v8h b1h = *(const v8h*)(&Bt[no + 16 + l15][16 + half * 8]);

        v16h af0 = __builtin_shufflevector(a0l, a0h, 0,1,2,3,4,5,6,7,8,9,10,11,12,13,14,15);
        v16h af1 = __builtin_shufflevector(a1l, a1h, 0,1,2,3,4,5,6,7,8,9,10,11,12,13,14,15);
        v16h bf0 = __builtin_shufflevector(b0l, b0h, 0,1,2,3,4,5,6,7,8,9,10,11,12,13,14,15);
        v16h bf1 = __builtin_shufflevector(b1l, b1h, 0,1,2,3,4,5,6,7,8,9,10,11,12,13,14,15);

        acc00 = __builtin_amdgcn_wmma_f32_16x16x32_f16(false, af0, false, bf0, (short)0, acc00, false, false);
        acc01 = __builtin_amdgcn_wmma_f32_16x16x32_f16(false, af0, false, bf1, (short)0, acc01, false, false);
        acc10 = __builtin_amdgcn_wmma_f32_16x16x32_f16(false, af1, false, bf0, (short)0, acc10, false, false);
        acc11 = __builtin_amdgcn_wmma_f32_16x16x32_f16(false, af1, false, bf1, (short)0, acc11, false, false);
        __syncthreads();
    }

    // Epilogue. C/D layout: VGPR r -> M = r + 8*half_group, N = lane&15.
    const int n0 = colBlk + no + l15;
    const int m0 = rowBlk + mo + half * 8;
    const float bv0 = bias[n0];
    const float bv1 = bias[n0 + 16];
    #pragma unroll
    for (int r = 0; r < 8; ++r) {
        float v00 = acc00[r] + bv0, v01 = acc01[r] + bv1;
        float v10 = acc10[r] + bv0, v11 = acc11[r] + bv1;
        if (RELU) {
            v00 = fmaxf(v00, 0.0f); v01 = fmaxf(v01, 0.0f);
            v10 = fmaxf(v10, 0.0f); v11 = fmaxf(v11, 0.0f);
        }
        const size_t r0 = (size_t)(m0 + r) * Nd;
        const size_t r1 = (size_t)(m0 + 16 + r) * Nd;
        C16[r0 + n0]      = (_Float16)v00;
        C16[r0 + n0 + 16] = (_Float16)v01;
        C16[r1 + n0]      = (_Float16)v10;
        C16[r1 + n0 + 16] = (_Float16)v11;
        if (WRITE_F32) {
            C32[r0 + n0]      = v00;
            C32[r0 + n0 + 16] = v01;
            C32[r1 + n0]      = v10;
            C32[r1 + n0 + 16] = v11;
        }
    }
}

// ---------------------------------------------------------------------------
// Elementwise / gather / reduction kernels
// ---------------------------------------------------------------------------
__global__ void cvt_f32_f16_kernel(const float* __restrict__ x,
                                   _Float16* __restrict__ y, size_t n)
{
    size_t t = (size_t)blockIdx.x * blockDim.x + threadIdx.x;
    if (t < n) y[t] = (_Float16)x[t];
}

// w[Kd x Nd] f32  ->  wt[Nd x Kd] f16   (weights pre-transposed for the GEMM)
__global__ void cvt_transpose_kernel(const float* __restrict__ w,
                                     _Float16* __restrict__ wt, int Kd, int Nd)
{
    int t = blockIdx.x * blockDim.x + threadIdx.x;
    if (t >= Kd * Nd) return;
    int k = t / Nd, n = t % Nd;
    wt[(size_t)n * Kd + k] = (_Float16)w[t];
}

__global__ void edge_masks_kernel(const int* __restrict__ Eidx,
                                  const float* __restrict__ mask,
                                  float* __restrict__ ma,
                                  float* __restrict__ bw)
{
    using namespace cfg;
    int e = blockIdx.x * blockDim.x + threadIdx.x;
    if (e >= NE) return;
    int node = e / K;
    int b = node / N;
    int n = node % N;
    int nb = Eidx[e];
    float mv = mask[node];
    ma[e] = mv * mask[b * N + nb];                 // mask_attend
    bw[e] = mv * ((nb - n < 0) ? 1.0f : 0.0f);     // mask_bw
}

__global__ void embed_S_kernel(const int* __restrict__ S,
                               const float* __restrict__ Ws,
                               _Float16* __restrict__ hS)
{
    using namespace cfg;
    int t = blockIdx.x * blockDim.x + threadIdx.x;
    if (t >= BN * H) return;
    int node = t >> 7;
    int h = t & (H - 1);
    hS[t] = (_Float16)Ws[S[node] * H + h];
}

// X[e] = [ h_V[n] | h_E[e] | h_V[E_idx] ]  (width 384)
__global__ void build_enc_x_kernel(const _Float16* __restrict__ hV,
                                   const _Float16* __restrict__ hE,
                                   const int* __restrict__ Eidx,
                                   _Float16* __restrict__ X)
{
    using namespace cfg;
    size_t t = (size_t)blockIdx.x * blockDim.x + threadIdx.x;
    if (t >= (size_t)NE * 384) return;
    size_t e = t / 384;
    int c = (int)(t % 384);
    int node = (int)(e / K);
    int b = node / N;
    _Float16 v;
    if (c < 128)      v = hV[(size_t)node * H + c];
    else if (c < 256) v = hE[e * H + (c - 128)];
    else              v = hV[((size_t)(b * N + Eidx[e])) * H + (c - 256)];
    X[e * 384 + c] = v;
}

// enc_fw[e] = mask_fw * [ h_E | 0 | gather(h_Vd0) ]  (width 384)
__global__ void build_encfw_kernel(const _Float16* __restrict__ hE,
                                   const _Float16* __restrict__ hVd0,
                                   const int* __restrict__ Eidx,
                                   const float* __restrict__ mask,
                                   _Float16* __restrict__ out)
{
    using namespace cfg;
    size_t t = (size_t)blockIdx.x * blockDim.x + threadIdx.x;
    if (t >= (size_t)NE * 384) return;
    size_t e = t / 384;
    int c = (int)(t % 384);
    int node = (int)(e / K);
    int b = node / N;
    int n = node % N;
    int nb = Eidx[e];
    float fw = mask[node] * ((nb - n < 0) ? 0.0f : 1.0f);
    float v = 0.0f;
    if (c < 128)       v = (float)hE[e * H + c];
    else if (c >= 256) v = (float)hVd0[((size_t)(b * N + nb)) * H + (c - 256)];
    out[e * 384 + c] = (_Float16)(fw * v);
}

// X[e] = [ h_Vd[n] | bw*(h_E | h_S[idx] | h_Vd[idx]) + enc_fw ]  (width 512)
__global__ void build_dec_x_kernel(const _Float16* __restrict__ hVd,
                                   const _Float16* __restrict__ hE,
                                   const _Float16* __restrict__ hS,
                                   const _Float16* __restrict__ encfw,
                                   const float* __restrict__ bw,
                                   const int* __restrict__ Eidx,
                                   _Float16* __restrict__ X)
{
    using namespace cfg;
    size_t t = (size_t)blockIdx.x * blockDim.x + threadIdx.x;
    if (t >= (size_t)NE * 512) return;
    size_t e = t / 512;
    int c = (int)(t % 512);
    int node = (int)(e / K);
    int b = node / N;
    float v;
    if (c < 128) {
        v = (float)hVd[(size_t)node * H + c];
    } else {
        int j = c - 128;
        int nb = Eidx[e];
        float base;
        if (j < 128)      base = (float)hE[e * H + j];
        else if (j < 256) base = (float)hS[((size_t)(b * N + nb)) * H + (j - 128)];
        else              base = (float)hVd[((size_t)(b * N + nb)) * H + (j - 256)];
        v = bw[e] * base + (float)encfw[e * 384 + j];
    }
    X[e * 512 + c] = (_Float16)v;
}

// dh[node,h] = (1/K) * sum_k ma[node,k] * m[node,k,h]
__global__ void edge_reduce_kernel(const _Float16* __restrict__ m,
                                   const float* __restrict__ ma,
                                   float* __restrict__ dh)
{
    using namespace cfg;
    int t = blockIdx.x * blockDim.x + threadIdx.x;
    if (t >= BN * H) return;
    int node = t >> 7;
    int h = t & (H - 1);
    float acc = 0.0f;
    for (int k = 0; k < K; ++k) {
        size_t e = (size_t)node * K + k;
        float w = ma ? ma[e] : 1.0f;
        acc += w * (float)m[e * H + h];
    }
    dh[t] = acc * (1.0f / (float)K);
}

// h = LN(hin + delta) * (maskV?); writes f32 master + f16 mirror
__global__ __launch_bounds__(128)
void residual_ln_kernel(const float* __restrict__ hin,
                        const float* __restrict__ delta,
                        const float* __restrict__ gamma,
                        const float* __restrict__ beta,
                        const float* __restrict__ maskV,
                        float* __restrict__ hout,
                        _Float16* __restrict__ hout16)
{
    using namespace cfg;
    const int node = blockIdx.x;
    const int h = threadIdx.x;
    __shared__ float red[128];
    float x = hin[(size_t)node * H + h] + delta[(size_t)node * H + h];
    red[h] = x;
    __syncthreads();
    for (int s = 64; s > 0; s >>= 1) { if (h < s) red[h] += red[h + s]; __syncthreads(); }
    float mean = red[0] * (1.0f / H);
    __syncthreads();
    float d = x - mean;
    red[h] = d * d;
    __syncthreads();
    for (int s = 64; s > 0; s >>= 1) { if (h < s) red[h] += red[h + s]; __syncthreads(); }
    float var = red[0] * (1.0f / H);
    float y = gamma[h] * d * rsqrtf(var + 1e-5f) + beta[h];
    if (maskV) y *= maskV[node];
    hout[(size_t)node * H + h] = y;
    hout16[(size_t)node * H + h] = (_Float16)y;
}

__global__ void musig_kernel(const float* __restrict__ hV32,
                             const float* __restrict__ wmu, const float* __restrict__ bmu,
                             const float* __restrict__ wsg, const float* __restrict__ bsg,
                             const float* __restrict__ eps,
                             float* __restrict__ omu, float* __restrict__ osig,
                             float* __restrict__ z)
{
    using namespace cfg;
    int t = blockIdx.x * blockDim.x + threadIdx.x;
    if (t >= B * Z) return;
    int b = t / Z, zc = t % Z;
    const float* hrow = hV32 + ((size_t)(b * N + (N - 1))) * H;
    float mu = bmu[zc], sg = bsg[zc];
    for (int h = 0; h < H; ++h) {
        float x = hrow[h];
        mu += x * wmu[h * Z + zc];
        sg += x * wsg[h * Z + zc];
    }
    omu[t] = mu;
    osig[t] = sg;
    z[t] = mu + eps[t] * expf(0.5f * sg);
}

// g_in[node] = [ z[b] | mean_k h_E | mean_k gather(h_Vsnap) ]  (width 320)
__global__ void build_gin_kernel(const float* __restrict__ z,
                                 const _Float16* __restrict__ hE,
                                 const _Float16* __restrict__ hVsnap,
                                 const int* __restrict__ Eidx,
                                 _Float16* __restrict__ gin)
{
    using namespace cfg;
    int t = blockIdx.x * blockDim.x + threadIdx.x;
    if (t >= BN * 320) return;
    int node = t / 320;
    int c = t % 320;
    int b = node / N;
    float v;
    if (c < 64) {
        v = z[b * Z + c];
    } else if (c < 192) {
        int cc = c - 64;
        float a = 0.0f;
        for (int k = 0; k < K; ++k) a += (float)hE[((size_t)node * K + k) * H + cc];
        v = a * (1.0f / (float)K);
    } else {
        int cc = c - 192;
        float a = 0.0f;
        for (int k = 0; k < K; ++k)
            a += (float)hVsnap[((size_t)(b * N + Eidx[node * K + k])) * H + cc];
        v = a * (1.0f / (float)K);
    }
    gin[(size_t)node * 320 + c] = (_Float16)v;
}

__global__ void mask_mul_kernel(_Float16* __restrict__ g, const float* __restrict__ mask)
{
    using namespace cfg;
    int t = blockIdx.x * blockDim.x + threadIdx.x;
    if (t >= BN * H) return;
    g[t] = (_Float16)((float)g[t] * mask[t >> 7]);
}

__global__ void recon_kernel(const _Float16* __restrict__ g,
                             const float* __restrict__ w, const float* __restrict__ b,
                             float* __restrict__ out)
{
    using namespace cfg;
    int t = blockIdx.x * blockDim.x + threadIdx.x;
    if (t >= BN * 3) return;
    int node = t / 3, j = t % 3;
    float p = b[j];
    for (int h = 0; h < H; ++h) p += (float)g[(size_t)node * H + h] * w[h * 3 + j];
    out[t] = atan2f(sinf(p), cosf(p));
}

__global__ void bcast_mu_kernel(const float* __restrict__ mu, _Float16* __restrict__ A)
{
    using namespace cfg;
    int t = blockIdx.x * blockDim.x + threadIdx.x;
    if (t >= BN * Z) return;
    int node = t / Z;
    A[t] = (_Float16)mu[(node / N) * Z + (t % Z)];
}

__global__ void logits_kernel(const _Float16* __restrict__ hVd,
                              const float* __restrict__ w, const float* __restrict__ b,
                              float* __restrict__ out)
{
    using namespace cfg;
    int node = blockIdx.x * blockDim.x + threadIdx.x;
    if (node >= BN) return;
    float lg[VOCAB];
    for (int v = 0; v < VOCAB; ++v) lg[v] = b[v];
    for (int h = 0; h < H; ++h) {
        float x = (float)hVd[(size_t)node * H + h];
        for (int v = 0; v < VOCAB; ++v) lg[v] += x * w[h * VOCAB + v];
    }
    float mx = lg[0];
    for (int v = 1; v < VOCAB; ++v) mx = fmaxf(mx, lg[v]);
    float s = 0.0f;
    for (int v = 0; v < VOCAB; ++v) s += expf(lg[v] - mx);
    float lse = mx + logf(s);
    for (int v = 0; v < VOCAB; ++v) out[(size_t)node * VOCAB + v] = lg[v] - lse;
}

// ---------------------------------------------------------------------------
// Host orchestration
// ---------------------------------------------------------------------------
static inline dim3 g1(size_t n, int bs = 256) { return dim3((unsigned)((n + bs - 1) / bs)); }

extern "C" void kernel_launch(void* const* d_in, const int* in_sizes, int n_in,
                              void* d_out, int out_size, void* d_ws, size_t ws_size,
                              hipStream_t stream)
{
    using namespace cfg;
    (void)in_sizes; (void)n_in; (void)out_size; (void)ws_size;

    auto Fp = [&](int i) -> const float* { return (const float*)d_in[i]; };
    auto Ip = [&](int i) -> const int*   { return (const int*)d_in[i]; };

    const float* V    = Fp(0);
    const float* E    = Fp(1);
    const int*   S    = Ip(2);
    const int*   Eidx = Ip(3);
    const float* mask = Fp(4);
    const float* eps  = Fp(5);

    // params: jax pytree leaves, dict keys sorted
    int pi = 6;
    const float* We_b  = Fp(pi++); const float* We_w  = Fp(pi++);   // W_e
    const float* Wo_b  = Fp(pi++); const float* Wo_w  = Fp(pi++);   // W_out
    const float* Ws    = Fp(pi++);                                  // W_s
    const float* Wv_b  = Fp(pi++); const float* Wv_w  = Fp(pi++);   // W_v
    struct Mpnn {
        const float *W1b,*W1w,*W2b,*W2w,*W3b,*W3w,*Winb,*Winw,*Wob,*Wow,
                    *ln1b,*ln1g,*ln2b,*ln2g;
    };
    Mpnn dec[3], enc[3];
    auto grab = [&](Mpnn& m) {
        m.W1b=Fp(pi++); m.W1w=Fp(pi++); m.W2b=Fp(pi++); m.W2w=Fp(pi++);
        m.W3b=Fp(pi++); m.W3w=Fp(pi++); m.Winb=Fp(pi++); m.Winw=Fp(pi++);
        m.Wob=Fp(pi++); m.Wow=Fp(pi++);
        m.ln1b=Fp(pi++); m.ln1g=Fp(pi++); m.ln2b=Fp(pi++); m.ln2g=Fp(pi++);
    };
    for (int l = 0; l < 3; ++l) grab(dec[l]);
    for (int l = 0; l < 3; ++l) grab(enc[l]);
    const float* fcd_b = Fp(pi++); const float* fcd_w = Fp(pi++);   // fc_decode
    const float* fmu_b = Fp(pi++); const float* fmu_w = Fp(pi++);   // fc_mu
    const float* fsg_b = Fp(pi++); const float* fsg_w = Fp(pi++);   // fc_sig
    const float* gen_b = Fp(pi++); const float* gen_w = Fp(pi++);   // gen
    const float* rec_b = Fp(pi++); const float* rec_w = Fp(pi++);   // recon_out

    // workspace bump allocator
    uintptr_t base = (uintptr_t)d_ws;
    size_t off = 0;
    auto b16 = [&](size_t n) -> _Float16* {
        off = (off + 255) & ~(size_t)255; _Float16* p = (_Float16*)(base + off); off += n * 2; return p;
    };
    auto b32 = [&](size_t n) -> float* {
        off = (off + 255) & ~(size_t)255; float* p = (float*)(base + off); off += n * 4; return p;
    };

    auto cvt = [&](const float* src, _Float16* dst, size_t n) {
        cvt_f32_f16_kernel<<<g1(n), dim3(256), 0, stream>>>(src, dst, n);
    };
    auto cvtT = [&](const float* w, _Float16* wt, int Kd, int Nd) {
        cvt_transpose_kernel<<<g1((size_t)Kd * Nd), dim3(256), 0, stream>>>(w, wt, Kd, Nd);
    };
    auto gemm = [&](const _Float16* A, const _Float16* WT, const float* bias,
                    _Float16* C16, float* C32, int M, int Kd, int Nd, bool relu) {
        dim3 grid(M / 64, Nd / 64), block(128);
        if (C32)       wmma_gemm_kernel<false, true ><<<grid, block, 0, stream>>>(A, WT, bias, C16, C32, M, Kd, Nd);
        else if (relu) wmma_gemm_kernel<true,  false><<<grid, block, 0, stream>>>(A, WT, bias, C16, nullptr, M, Kd, Nd);
        else           wmma_gemm_kernel<false, false><<<grid, block, 0, stream>>>(A, WT, bias, C16, nullptr, M, Kd, Nd);
    };

    // f16 transposed weight mirrors (WT[Nd][Kd])
    _Float16* wv16  = b16((size_t)H * H);
    _Float16* we16  = b16((size_t)H * H);
    _Float16 *eW1[3], *eW2[3], *eW3[3], *eWin[3], *eWo[3];
    _Float16 *dW1[3], *dW2[3], *dW3[3], *dWin[3], *dWo[3];
    for (int l = 0; l < 3; ++l) {
        eW1[l] = b16((size_t)384 * H); eW2[l] = b16((size_t)H * H); eW3[l] = b16((size_t)H * H);
        eWin[l] = b16((size_t)H * FF); eWo[l] = b16((size_t)FF * H);
        dW1[l] = b16((size_t)512 * H); dW2[l] = b16((size_t)H * H); dW3[l] = b16((size_t)H * H);
        dWin[l] = b16((size_t)H * FF); dWo[l] = b16((size_t)FF * H);
    }
    _Float16* fcd16 = b16((size_t)Z * H);
    _Float16* gen16 = b16((size_t)320 * H);

    // activations
    _Float16* V16     = b16((size_t)BN * H);
    _Float16* E16     = b16((size_t)NE * H);
    _Float16* hV16    = b16((size_t)BN * H);
    float*    hV32    = b32((size_t)BN * H);
    _Float16* hVsnap  = b16((size_t)BN * H);
    _Float16* hE16    = b16((size_t)NE * H);
    _Float16* hS16    = b16((size_t)BN * H);
    _Float16* hVd16   = b16((size_t)BN * H);
    float*    hVd32   = b32((size_t)BN * H);
    _Float16* X16     = b16((size_t)NE * 512);
    _Float16* mA16    = b16((size_t)NE * H);
    _Float16* mB16    = b16((size_t)NE * H);
    _Float16* encfw   = b16((size_t)NE * 384);
    _Float16* ff16    = b16((size_t)BN * FF);
    _Float16* tmp16   = b16((size_t)BN * H);
    _Float16* ginA16  = b16((size_t)BN * 320);
    _Float16* g16     = b16((size_t)BN * H);
    _Float16* muA16   = b16((size_t)BN * Z);
    float*    ma32    = b32(NE);
    float*    bw32    = b32(NE);
    float*    dh32    = b32((size_t)BN * H);
    float*    z32     = b32(B * Z);

    float* out     = (float*)d_out;
    float* o_recon = out;                          // BN*3
    float* o_lp    = out + (size_t)BN * 3;         // BN*VOCAB
    float* o_mu    = o_lp + (size_t)BN * VOCAB;    // B*Z
    float* o_sig   = o_mu + (size_t)B * Z;         // B*Z

    // ---- weight conversions (f32 [Kd,Nd] -> f16 transposed [Nd,Kd]) ----
    cvtT(Wv_w, wv16, H, H);
    cvtT(We_w, we16, H, H);
    for (int l = 0; l < 3; ++l) {
        cvtT(enc[l].W1w, eW1[l], 384, H);
        cvtT(enc[l].W2w, eW2[l], H, H);
        cvtT(enc[l].W3w, eW3[l], H, H);
        cvtT(enc[l].Winw, eWin[l], H, FF);
        cvtT(enc[l].Wow, eWo[l], FF, H);
        cvtT(dec[l].W1w, dW1[l], 512, H);
        cvtT(dec[l].W2w, dW2[l], H, H);
        cvtT(dec[l].W3w, dW3[l], H, H);
        cvtT(dec[l].Winw, dWin[l], H, FF);
        cvtT(dec[l].Wow, dWo[l], FF, H);
    }
    cvtT(fcd_w, fcd16, Z, H);
    cvtT(gen_w, gen16, 320, H);

    // ---- input projections ----
    cvt(V, V16, (size_t)BN * H);
    cvt(E, E16, (size_t)NE * H);
    gemm(V16, wv16, Wv_b, hV16, hV32, BN, H, H, false);          // h_V = V @ W_v
    gemm(E16, we16, We_b, hE16, nullptr, NE, H, H, false);       // h_E = E @ W_e
    edge_masks_kernel<<<g1(NE), 256, 0, stream>>>(Eidx, mask, ma32, bw32);
    embed_S_kernel<<<g1((size_t)BN * H), 256, 0, stream>>>(S, Ws, hS16);

    // ---- encoder ----
    for (int l = 0; l < 3; ++l) {
        if (l == 2)  // snapshot h_V feeding last-layer h_EV (needed for g_in mean)
            hipMemcpyAsync(hVsnap, hV16, (size_t)BN * H * sizeof(_Float16),
                           hipMemcpyDeviceToDevice, stream);
        build_enc_x_kernel<<<g1((size_t)NE * 384), 256, 0, stream>>>(hV16, hE16, Eidx, X16);
        gemm(X16, eW1[l], enc[l].W1b, mA16, nullptr, NE, 384, H, true);
        gemm(mA16, eW2[l], enc[l].W2b, mB16, nullptr, NE, H, H, true);
        gemm(mB16, eW3[l], enc[l].W3b, mA16, nullptr, NE, H, H, false);
        edge_reduce_kernel<<<g1((size_t)BN * H), 256, 0, stream>>>(mA16, ma32, dh32);
        residual_ln_kernel<<<BN, 128, 0, stream>>>(hV32, dh32, enc[l].ln1g, enc[l].ln1b,
                                                   nullptr, hV32, hV16);
        gemm(hV16, eWin[l], enc[l].Winb, ff16, nullptr, BN, H, FF, true);
        gemm(ff16, eWo[l], enc[l].Wob, tmp16, dh32, BN, FF, H, false);
        residual_ln_kernel<<<BN, 128, 0, stream>>>(hV32, dh32, enc[l].ln2g, enc[l].ln2b,
                                                   mask, hV32, hV16);
    }

    // ---- latent ----
    musig_kernel<<<g1(B * Z), 256, 0, stream>>>(hV32, fmu_w, fmu_b, fsg_w, fsg_b, eps,
                                                o_mu, o_sig, z32);

    // ---- generator / reconstruction ----
    build_gin_kernel<<<g1((size_t)BN * 320), 256, 0, stream>>>(z32, hE16, hVsnap, Eidx, ginA16);
    gemm(ginA16, gen16, gen_b, g16, nullptr, BN, 320, H, true);
    mask_mul_kernel<<<g1((size_t)BN * H), 256, 0, stream>>>(g16, mask);
    recon_kernel<<<g1((size_t)BN * 3), 256, 0, stream>>>(g16, rec_w, rec_b, o_recon);

    // ---- decoder init ----
    bcast_mu_kernel<<<g1((size_t)BN * Z), 256, 0, stream>>>(o_mu, muA16);
    gemm(muA16, fcd16, fcd_b, hVd16, hVd32, BN, Z, H, false);    // h_Vd = mu @ fc_decode
    build_encfw_kernel<<<g1((size_t)NE * 384), 256, 0, stream>>>(hE16, hVd16, Eidx, mask, encfw);

    // ---- decoder ----
    for (int l = 0; l < 3; ++l) {
        build_dec_x_kernel<<<g1((size_t)NE * 512), 256, 0, stream>>>(
            hVd16, hE16, hS16, encfw, bw32, Eidx, X16);
        gemm(X16, dW1[l], dec[l].W1b, mA16, nullptr, NE, 512, H, true);
        gemm(mA16, dW2[l], dec[l].W2b, mB16, nullptr, NE, H, H, true);
        gemm(mB16, dW3[l], dec[l].W3b, mA16, nullptr, NE, H, H, false);
        edge_reduce_kernel<<<g1((size_t)BN * H), 256, 0, stream>>>(mA16, nullptr, dh32);
        residual_ln_kernel<<<BN, 128, 0, stream>>>(hVd32, dh32, dec[l].ln1g, dec[l].ln1b,
                                                   nullptr, hVd32, hVd16);
        gemm(hVd16, dWin[l], dec[l].Winb, ff16, nullptr, BN, H, FF, true);
        gemm(ff16, dWo[l], dec[l].Wob, tmp16, dh32, BN, FF, H, false);
        residual_ln_kernel<<<BN, 128, 0, stream>>>(hVd32, dh32, dec[l].ln2g, dec[l].ln2b,
                                                   mask, hVd32, hVd16);
    }

    // ---- output head ----
    logits_kernel<<<g1(BN, 128), 128, 0, stream>>>(hVd16, Wo_w, Wo_b, o_lp);
}
